// GAU_Attention_20813411516822
// MI455X (gfx1250) — matrix operand order repaired and checked by reading, not compiled, not run on hardware
//
#include <hip/hip_runtime.h>
#include <hip/hip_bf16.h>
#include <stdint.h>

// ---------------------------------------------------------------------------
// GAU attention for MI455X (gfx1250): bf16 WMMA + async-to-LDS staging.
//   HIDDEN=768, HEADS=12, HEAD_DIM=64, B=4, S=2048, S_REL=128
// ---------------------------------------------------------------------------

typedef __attribute__((ext_vector_type(16))) __bf16 v16bf;
typedef __attribute__((ext_vector_type(8)))  __bf16 bf16x8;
typedef __attribute__((ext_vector_type(8)))  float  v8f;
typedef unsigned short u16;

#define HIDDEN 768
#define HEADS  12
#define HDIM   64
#define BATCH  4
#define SEQ    2048

__device__ __forceinline__ u16 f2bf(float f) {
    unsigned u = __float_as_uint(f);
    unsigned r = u + 0x7FFFu + ((u >> 16) & 1u);   // RNE
    return (u16)(r >> 16);
}

__device__ __forceinline__ v16bf mk_frag(bf16x8 lo, bf16x8 hi) {
    v16bf r;
#pragma unroll
    for (int i = 0; i < 8; ++i) { r[i] = lo[i]; r[8 + i] = hi[i]; }
    return r;
}
__device__ __forceinline__ v16bf load_frag(const u16* p0, const u16* p1) {
    return mk_frag(*reinterpret_cast<const bf16x8*>(p0),
                   *reinterpret_cast<const bf16x8*>(p1));
}
__device__ __forceinline__ v8f wmma_bf16(v16bf a, v16bf b, v8f c) {
    return __builtin_amdgcn_wmma_f32_16x16x32_bf16(false, a, false, b,
                                                   (short)0, c, false, false);
}

// async copy of 16 bytes / lane: global -> LDS, tracked by ASYNCcnt
__device__ __forceinline__ void async_b128(unsigned lds_addr, const u16* gptr) {
    unsigned long ga = (unsigned long)(uintptr_t)gptr;
    asm volatile("global_load_async_to_lds_b128 %0, %1, off"
                 :: "v"(lds_addr), "v"(ga) : "memory");
}
__device__ __forceinline__ void wait_async0() {
    asm volatile("s_wait_asynccnt 0" ::: "memory");
}
__device__ __forceinline__ void wait_ds0() {
    asm volatile("s_wait_dscnt 0" ::: "memory");
}

// ---------------------------------------------------------------------------
// f32 -> bf16 (same layout)
// ---------------------------------------------------------------------------
__global__ void cvt_bf16_kernel(const float* __restrict__ in,
                                u16* __restrict__ out, long n) {
    long i = (long)blockIdx.x * blockDim.x + threadIdx.x;
    if (i < n) out[i] = f2bf(in[i]);
}

// f32 [K,N] -> bf16 [N,K]  (weights become ready-to-use WMMA B operands)
__global__ void cvt_tr_bf16_kernel(const float* __restrict__ in,
                                   u16* __restrict__ out, int K, int N) {
    long i = (long)blockIdx.x * blockDim.x + threadIdx.x;
    if (i >= (long)K * N) return;
    int n = (int)(i % N), k = (int)(i / N);
    out[(long)n * K + k] = f2bf(in[i]);
}

// ---------------------------------------------------------------------------
// RoPE of broadcast a_param/b_param -> bf16 [2048,128]
// ---------------------------------------------------------------------------
__global__ void rope_kernel(const float* __restrict__ ap,
                            const float* __restrict__ bp,
                            u16* __restrict__ ar, u16* __restrict__ br) {
    int s = blockIdx.x;              // 0..2047
    int k = threadIdx.x;             // 0..127
    int kk = k & 63;
    float inv_freq = __powf(10000.0f, -(float)kk / 64.0f);
    float ang = (float)s * inv_freq;
    float sn = __sinf(ang), cs = __cosf(ang);
    float a1 = ap[kk], a2 = ap[64 + kk];
    float b1 = bp[kk], b2 = bp[64 + kk];
    float va = (k < 64) ? (a1 * cs - a2 * sn) : (a2 * cs + a1 * sn);
    float vb = (k < 64) ? (b1 * cs - b2 * sn) : (b2 * cs + b1 * sn);
    ar[(long)s * 128 + k] = f2bf(va);
    br[(long)s * 128 + k] = f2bf(vb);
}

// ---------------------------------------------------------------------------
// Generic bf16 WMMA GEMM:  C[M,N] = A[M,K] @ Bt[N,K]^T (+bias)
// Each wave computes a 16x64 strip (A fragment reused across 4 N-tiles).
// mode 0: bf16 row-major out  | mode 1: f32 row-major out
// mode 2: bf16 out scattered as Vt[B][H][D][S] (V transposed for attention)
// ---------------------------------------------------------------------------
__global__ void gemm_bf16_wmma(const u16* __restrict__ A,
                               const u16* __restrict__ Bt,
                               const float* __restrict__ bias,
                               u16* __restrict__ outbf,
                               float* __restrict__ outf,
                               int M, int N, int K, int mode) {
    int lane = threadIdx.x & 31;
    int wave = threadIdx.x >> 5;
    int half = lane >> 4;
    int l16  = lane & 15;

    int tiles_n4 = N >> 6;                       // strips of 64 columns
    int strips   = (M >> 4) * tiles_n4;
    int strip    = blockIdx.x * 8 + wave;
    if (strip >= strips) return;
    int mt = strip / tiles_n4;
    int n4 = strip % tiles_n4;

    const u16* Ar = A + (long)(mt * 16 + l16) * K;         // A row (M = lane%16)
    const u16* Br0 = Bt + (long)(n4 * 64 +  0 + l16) * K;  // B rows (N = lane%16)
    const u16* Br1 = Bt + (long)(n4 * 64 + 16 + l16) * K;
    const u16* Br2 = Bt + (long)(n4 * 64 + 32 + l16) * K;
    const u16* Br3 = Bt + (long)(n4 * 64 + 48 + l16) * K;

    int hk = half * 16;   // A: chunk K-offset per half
    int bk = half * 16;   // B: chunk K-offset per half

    v8f c0 = {}, c1 = {}, c2 = {}, c3 = {};
    for (int k0 = 0; k0 < K; k0 += 32) {
        v16bf a  = load_frag(Ar + k0 + hk,      Ar + k0 + hk + 8);
        v16bf b0 = load_frag(Br0 + k0 + bk,     Br0 + k0 + bk + 8);
        v16bf b1 = load_frag(Br1 + k0 + bk,     Br1 + k0 + bk + 8);
        v16bf b2 = load_frag(Br2 + k0 + bk,     Br2 + k0 + bk + 8);
        v16bf b3 = load_frag(Br3 + k0 + bk,     Br3 + k0 + bk + 8);
        c0 = wmma_bf16(a, b0, c0);
        c1 = wmma_bf16(a, b1, c1);
        c2 = wmma_bf16(a, b2, c2);
        c3 = wmma_bf16(a, b3, c3);
    }

    // epilogue: C layout -> element r has M = r + 8*half, N = lane%16
#pragma unroll
    for (int t = 0; t < 4; ++t) {
        v8f c = (t == 0) ? c0 : (t == 1) ? c1 : (t == 2) ? c2 : c3;
        int gc = n4 * 64 + t * 16 + l16;
        float bv = bias ? bias[gc] : 0.0f;
#pragma unroll
        for (int r = 0; r < 8; ++r) {
            int gm = mt * 16 + r + 8 * half;
            float v = c[r] + bv;
            if (mode == 0) {
                outbf[(long)gm * N + gc] = f2bf(v);
            } else if (mode == 1) {
                outf[(long)gm * N + gc] = v;
            } else {
                // V transposed: token row gm = b*2048 + s; col gc = h*64 + d
                int b = gm >> 11, s = gm & 2047;
                int h = gc >> 6,  d = gc & 63;
                outbf[(((long)(b * HEADS + h) << 6) + d) * SEQ + s] = f2bf(v);
            }
        }
    }
}

// ---------------------------------------------------------------------------
// Fused flash attention, WMMA + online softmax of squared-relu weights.
// K/V tiles are staged into LDS with double-buffered async-to-LDS copies
// shared by all 4 waves of the block (4x global-traffic reduction; the copy
// of block j+1 overlaps the WMMA compute of block j).
//   Q  : bf16 [B,S,HIDDEN]      relb: f32 [2048,2048]
//   Kb : bf16 [B,S,HIDDEN]      ctx : bf16 [B,S,HIDDEN]
//   Vt : bf16 [B,H,DHEAD,S]
// grid = B*H*(S/64), block = 128 (4 waves, 16 query rows per wave)
// ---------------------------------------------------------------------------
__global__ void flash_attn_wmma(const u16* __restrict__ Q,
                                const u16* __restrict__ Kb,
                                const u16* __restrict__ Vt,
                                const float* __restrict__ relb,
                                u16* __restrict__ ctx) {
    int lane = threadIdx.x & 31;
    int wave = threadIdx.x >> 5;
    int half = lane >> 4;
    int l16  = lane & 15;

    int bid = blockIdx.x;
    int qt   = bid & 31;                 // 32 query tiles of 64
    int rest = bid >> 5;
    int h = rest % HEADS;
    int b = rest / HEADS;
    int q0 = qt * 64 + wave * 16;

    const float inv8 = 0.125f;           // 1/sqrt(HEAD_DIM)
    const float invS = 1.0f / (float)SEQ;

    // Q A-fragments for this wave's 16 rows (M = lane%16), hoisted
    const u16* qp = Q + ((long)(b * SEQ + q0 + l16)) * HIDDEN + h * HDIM;
    int hk = half * 16;
    v16bf qa0 = load_frag(qp + 0  + hk, qp + 8 + hk);    // d = 0..31
    v16bf qa1 = load_frag(qp + 32 + hk, qp + 40 + hk);   // d = 32..63

    const u16* kbase = Kb + ((long)b * SEQ) * HIDDEN + h * HDIM;
    const u16* vbase = Vt + ((long)(b * HEADS + h)) * HDIM * SEQ;
    const float* bias_row = relb + (long)(q0 + 8 * half) * SEQ; // + r*SEQ

    // LDS: double-buffered K (32 keys x 64 d) and V (64 d x 32 keys) tiles,
    // plus per-wave P staging tiles.  8KB + 8KB + 4KB = 20KB.
    __shared__ __align__(16) u16 sK[2][32 * 64];
    __shared__ __align__(16) u16 sV[2][64 * 32];
    __shared__ __align__(16) u16 pb[4][16 * 32];
    u16* myp = pb[wave];

    // ---- async tile fetch: each wave copies its share of K (8 rows of
    // 128B, 8 lanes/row) and V (16 rows of 64B, 4 lanes/row) ----
    auto issue_tile = [&](int buf, int j) {
#pragma unroll
        for (int i = 0; i < 2; ++i) {
            int krow = wave * 8 + i * 4 + (lane >> 3);         // 0..31
            const u16* kg = kbase + (long)(j + krow) * HIDDEN + (lane & 7) * 8;
            unsigned kl = (unsigned)(uintptr_t)&sK[buf][krow * 64 + (lane & 7) * 8];
            async_b128(kl, kg);
            int vrow = wave * 16 + i * 8 + (lane >> 2);        // 0..63 (d)
            const u16* vg = vbase + (long)vrow * SEQ + j + (lane & 3) * 8;
            unsigned vl = (unsigned)(uintptr_t)&sV[buf][vrow * 32 + (lane & 3) * 8];
            async_b128(vl, vg);
        }
    };

    v8f o0 = {}, o1 = {}, o2 = {}, o3 = {};
    float m_i[8], l_i[8];
#pragma unroll
    for (int r = 0; r < 8; ++r) { m_i[r] = 0.0f; l_i[r] = 0.0f; }

    issue_tile(0, 0);
    wait_async0();
    __syncthreads();

    int buf = 0;
    for (int j = 0; j < SEQ; j += 32) {
        if (j + 32 < SEQ) issue_tile(buf ^ 1, j + 32);

        // ---- scores: S[16 x 32] = Q(16x64) @ K^T, two 16-key N-tiles ----
        // B frag: lane's key row = l16 (+16), d-chunk at c*32 + half*16
        const u16* k0p = &sK[buf][(l16)      * 64];
        const u16* k1p = &sK[buf][(16 + l16) * 64];
        int bk = half * 16;
        v8f s0 = {}, s1 = {};
        {
            v16bf kb = load_frag(k0p + bk,      k0p + bk + 8);        // d 0..31
            s0 = wmma_bf16(qa0, kb, s0);
            kb = load_frag(k0p + 32 + bk,  k0p + 32 + bk + 8);        // d 32..63
            s0 = wmma_bf16(qa1, kb, s0);
            kb = load_frag(k1p + bk,       k1p + bk + 8);
            s1 = wmma_bf16(qa0, kb, s1);
            kb = load_frag(k1p + 32 + bk,  k1p + 32 + bk + 8);
            s1 = wmma_bf16(qa1, kb, s1);
        }

        // ---- bias + squared-relu; row-max over 16 lanes of each half ----
        float w0[8], w1[8], mx[8];
#pragma unroll
        for (int r = 0; r < 8; ++r) {
            float bi0 = bias_row[(long)r * SEQ + j + l16];
            float bi1 = bias_row[(long)r * SEQ + j + 16 + l16];
            float t0 = (s0[r] * inv8 + bi0) * invS;
            float t1 = (s1[r] * inv8 + bi1) * invS;
            t0 = fmaxf(t0, 0.0f);  t1 = fmaxf(t1, 0.0f);
            w0[r] = t0 * t0;       w1[r] = t1 * t1;
            mx[r] = fmaxf(w0[r], w1[r]);
        }
#pragma unroll
        for (int r = 0; r < 8; ++r) {
            mx[r] = fmaxf(mx[r], __shfl_xor(mx[r], 1, 32));
            mx[r] = fmaxf(mx[r], __shfl_xor(mx[r], 2, 32));
            mx[r] = fmaxf(mx[r], __shfl_xor(mx[r], 4, 32));
            mx[r] = fmaxf(mx[r], __shfl_xor(mx[r], 8, 32));
        }

        // ---- online softmax update ----
        float p0[8], p1[8], rs[8];
#pragma unroll
        for (int r = 0; r < 8; ++r) {
            float mn = fmaxf(m_i[r], mx[r]);
            float sc = __expf(m_i[r] - mn);
            p0[r] = __expf(w0[r] - mn);
            p1[r] = __expf(w1[r] - mn);
            rs[r] = p0[r] + p1[r];
            m_i[r] = mn;
            l_i[r] *= sc;
            o0[r] *= sc; o1[r] *= sc; o2[r] *= sc; o3[r] *= sc;
        }
#pragma unroll
        for (int r = 0; r < 8; ++r) {
            rs[r] += __shfl_xor(rs[r], 1, 32);
            rs[r] += __shfl_xor(rs[r], 2, 32);
            rs[r] += __shfl_xor(rs[r], 4, 32);
            rs[r] += __shfl_xor(rs[r], 8, 32);
            l_i[r] += rs[r];
        }

        // ---- stage P (16x32) through LDS: C layout -> A layout ----
#pragma unroll
        for (int r = 0; r < 8; ++r) {
            int row = r + 8 * half;
            myp[row * 32 + l16]      = f2bf(p0[r]);
            myp[row * 32 + 16 + l16] = f2bf(p1[r]);
        }
        wait_ds0();
        v16bf pa = load_frag(myp + l16 * 32 + hk, myp + l16 * 32 + 8 + hk);

        // ---- O += P @ V_block : 4 d-tiles of 16 ----
        const u16* v0 = &sV[buf][( 0 + l16) * 32 + bk];
        const u16* v1 = &sV[buf][(16 + l16) * 32 + bk];
        const u16* v2 = &sV[buf][(32 + l16) * 32 + bk];
        const u16* v3 = &sV[buf][(48 + l16) * 32 + bk];
        o0 = wmma_bf16(pa, load_frag(v0, v0 + 8), o0);
        o1 = wmma_bf16(pa, load_frag(v1, v1 + 8), o1);
        o2 = wmma_bf16(pa, load_frag(v2, v2 + 8), o2);
        o3 = wmma_bf16(pa, load_frag(v3, v3 + 8), o3);

        wait_async0();        // next tile landed
        __syncthreads();      // visible to all waves; current tile free
        buf ^= 1;
    }

    // ---- normalize and store ctx (bf16, [B,S,HIDDEN]) ----
#pragma unroll
    for (int r = 0; r < 8; ++r) {
        float inv = 1.0f / l_i[r];
        int gm = q0 + r + 8 * half;
        long base = ((long)(b * SEQ + gm)) * HIDDEN + h * HDIM + l16;
        ctx[base +  0] = f2bf(o0[r] * inv);
        ctx[base + 16] = f2bf(o1[r] * inv);
        ctx[base + 32] = f2bf(o2[r] * inv);
        ctx[base + 48] = f2bf(o3[r] * inv);
    }
}

// ---------------------------------------------------------------------------
// Host-side orchestration
// ---------------------------------------------------------------------------
extern "C" void kernel_launch(void* const* d_in, const int* in_sizes, int n_in,
                              void* d_out, int out_size, void* d_ws, size_t ws_size,
                              hipStream_t stream) {
    const float* H  = (const float*)d_in[0];
    const float* Wq = (const float*)d_in[1];
    const float* bq = (const float*)d_in[2];
    const float* Wk = (const float*)d_in[3];
    const float* bk = (const float*)d_in[4];
    const float* Wv = (const float*)d_in[5];
    const float* bv = (const float*)d_in[6];
    const float* Wo = (const float*)d_in[7];
    const float* bo = (const float*)d_in[8];
    const float* ap = (const float*)d_in[9];
    const float* bp = (const float*)d_in[10];
    float* out = (float*)d_out;

    const long TOK = (long)BATCH * SEQ;            // 8192
    char* w = (char*)d_ws;
    u16* Hbf  = (u16*)w;                w += TOK * HIDDEN * 2;
    u16* WqT  = (u16*)w;                w += (long)HIDDEN * HIDDEN * 2;
    u16* WkT  = (u16*)w;                w += (long)HIDDEN * HIDDEN * 2;
    u16* WvT  = (u16*)w;                w += (long)HIDDEN * HIDDEN * 2;
    u16* WoT  = (u16*)w;                w += (long)HIDDEN * HIDDEN * 2;
    u16* Qbf  = (u16*)w;                w += TOK * HIDDEN * 2;
    u16* Kbf  = (u16*)w;                w += TOK * HIDDEN * 2;
    u16* Vt   = (u16*)w;                w += TOK * HIDDEN * 2;
    u16* Ctx  = (u16*)w;                w += TOK * HIDDEN * 2;
    u16* Arp  = (u16*)w;                w += (long)SEQ * 128 * 2;
    u16* Brp  = (u16*)w;                w += (long)SEQ * 128 * 2;
    float* Rel = (float*)w;             w += (long)SEQ * SEQ * 4;

    // 1) precision conversion
    {
        long n = TOK * HIDDEN;
        cvt_bf16_kernel<<<(int)((n + 255) / 256), 256, 0, stream>>>(H, Hbf, n);
        int g = (HIDDEN * HIDDEN + 255) / 256;
        cvt_tr_bf16_kernel<<<g, 256, 0, stream>>>(Wq, WqT, HIDDEN, HIDDEN);
        cvt_tr_bf16_kernel<<<g, 256, 0, stream>>>(Wk, WkT, HIDDEN, HIDDEN);
        cvt_tr_bf16_kernel<<<g, 256, 0, stream>>>(Wv, WvT, HIDDEN, HIDDEN);
        cvt_tr_bf16_kernel<<<g, 256, 0, stream>>>(Wo, WoT, HIDDEN, HIDDEN);
    }

    // 2) Q/K/V projections (WMMA GEMM). 8192x768x768 each.
    {
        int strips = ((int)TOK / 16) * (HIDDEN / 64);     // 6144
        int blocks = (strips + 7) / 8;
        gemm_bf16_wmma<<<blocks, 256, 0, stream>>>(Hbf, WqT, bq, Qbf, nullptr,
                                                   (int)TOK, HIDDEN, HIDDEN, 0);
        gemm_bf16_wmma<<<blocks, 256, 0, stream>>>(Hbf, WkT, bk, Kbf, nullptr,
                                                   (int)TOK, HIDDEN, HIDDEN, 0);
        gemm_bf16_wmma<<<blocks, 256, 0, stream>>>(Hbf, WvT, bv, Vt, nullptr,
                                                   (int)TOK, HIDDEN, HIDDEN, 2);
    }

    // 3) RoPE + relative bias GEMM (2048x2048, K=128)
    rope_kernel<<<SEQ, 128, 0, stream>>>(ap, bp, Arp, Brp);
    {
        int strips = (SEQ / 16) * (SEQ / 64);             // 4096
        int blocks = (strips + 7) / 8;
        gemm_bf16_wmma<<<blocks, 256, 0, stream>>>(Arp, Brp, nullptr, nullptr, Rel,
                                                   SEQ, SEQ, 128, 1);
    }

    // 4) fused flash attention (async-LDS double buffered)
    flash_attn_wmma<<<BATCH * HEADS * (SEQ / 64), 128, 0, stream>>>(Qbf, Kbf, Vt,
                                                                    Rel, Ctx);

    // 5) output projection -> f32 d_out
    {
        int strips = ((int)TOK / 16) * (HIDDEN / 64);
        int blocks = (strips + 7) / 8;
        gemm_bf16_wmma<<<blocks, 256, 0, stream>>>(Ctx, WoT, bo, nullptr, out,
                                                   (int)TOK, HIDDEN, HIDDEN, 1);
    }
}
